// RGTLayer_13649406067308
// MI455X (gfx1250) — matrix operand order repaired and checked by reading, not compiled, not run on hardware
//
#include <hip/hip_runtime.h>
#include <hip/hip_bf16.h>

// ---------------- problem constants (match reference) ----------------
#define NN   50000    // nodes  (50000 = 3125 * 16, no WMMA tail)
#define TT   3        // edge types
#define EE   800000   // edges per type
#define DD   128      // in_size
#define HH   2        // attention heads
#define CC   128      // out channels
#define HCC  256      // H*C
#define HIDD 128      // semantic hidden
#define SS   2        // semantic heads
#define NT   (NN*TT)  // 150000 = 9375 * 16

typedef __bf16 bf16_t;
typedef __attribute__((ext_vector_type(16))) __bf16 v16bf;
typedef __attribute__((ext_vector_type(8)))  float  v8f;

// ---------------- helpers ----------------
__device__ __forceinline__ unsigned fenc(float f) {
    int i = __float_as_int(f);
    return (i >= 0) ? ((unsigned)i | 0x80000000u) : (unsigned)(~i);
}
__device__ __forceinline__ float fdec(unsigned u) {
    return (u & 0x80000000u) ? __int_as_float((int)(u & 0x7fffffffu))
                             : __int_as_float(~(int)u);
}

// A-fragment (16x32 bf16, ISA 7.12.2): lane L = row (L&15), half (L>>4):
//   elems 0..7  = K { k + half*8 .. +7 },  elems 8..15 = K { k + half*8 + 16 .. +23 }
// p must point at A[row][k + half*8], 16B-aligned -> two global_load_b128.
__device__ __forceinline__ v16bf load_a_frag16(const bf16_t* __restrict__ p) {
    struct U32x8 { uint4 a, b; };
    const uint4* p4 = (const uint4*)p;
    U32x8 u{p4[0], p4[2]};
    return __builtin_bit_cast(v16bf, u);
}

// ---------------- generic kernels ----------------
__global__ void cvt_bf16_kernel(const float* __restrict__ in, bf16_t* __restrict__ out, long n) {
    long i = (long)blockIdx.x * blockDim.x + threadIdx.x;
    if (i < n) out[i] = (bf16_t)in[i];
}
__global__ void fill_zero_kernel(unsigned* __restrict__ p, long n) {
    long i = (long)blockIdx.x * blockDim.x + threadIdx.x;
    if (i < n) p[i] = 0u;
}

// Repack a weight matrix W (K x F, f32, row-major) into WMMA B-fragment order:
//   P[(((ct*ktiles + kt)*32 + lane)*16 + e] = (bf16) W[kt*32 + (lane>>4)*8 + (e&7) + (e>>3)*16][ct*16 + (lane&15)]
// so the GEMM reads one contiguous v16bf (32B) per lane per fragment.
__global__ void repack_b_kernel(const float* __restrict__ W, bf16_t* __restrict__ P,
                                int K, int F) {
    long i = (long)blockIdx.x * blockDim.x + threadIdx.x;
    long total = (long)K * F;
    if (i >= total) return;
    int e    = (int)(i & 15);
    long r   = i >> 4;
    int lane = (int)(r & 31);
    r >>= 5;
    int ktiles = K >> 5;
    int kt = (int)(r % ktiles);
    int ct = (int)(r / ktiles);
    int n = ct * 16 + (lane & 15);
    int k = kt * 32 + (lane >> 4) * 8 + (e & 7) + ((e >> 3) << 4);
    P[i] = (bf16_t)W[(size_t)k * F + n];
}

// ---------------- WMMA GEMM:  C = epi(A(MxK) @ B(KxF) + bias) ----------------
// Wave computes a 16x64 strip: 4 col-tiles share each A fragment (4x A reuse).
enum { EPI_F32 = 0, EPI_BF16 = 1, EPI_SIGMOID = 2 };

template <int EPI>
__device__ __forceinline__ void store_tile(v8f acc, int row0, int col0, int lane,
                                           const float* __restrict__ bias,
                                           float* __restrict__ Cf, bf16_t* __restrict__ Cb,
                                           int ldc) {
    int m = lane & 15, half = lane >> 4;
    int col = col0 + m;
    float bc = bias ? bias[col] : 0.f;
#pragma unroll
    for (int j = 0; j < 8; ++j) {
        int row = row0 + j + half * 8;
        float v = acc[j] + bc;
        if (EPI == EPI_SIGMOID) v = 1.f / (1.f + __expf(-v));
        if (EPI == EPI_BF16) Cb[(size_t)row * ldc + col] = (bf16_t)v;
        else                 Cf[(size_t)row * ldc + col] = v;
    }
}

template <int EPI>
__global__ void wmma_gemm_kernel(const bf16_t* __restrict__ A, int lda,
                                 const bf16_t* __restrict__ Bp,   // packed fragments
                                 const float* __restrict__ bias,
                                 float* __restrict__ Cf, bf16_t* __restrict__ Cb, int ldc,
                                 int M, int F, int Kdim) {
    int wid  = (int)((blockIdx.x * blockDim.x + threadIdx.x) >> 5);
    int lane = threadIdx.x & 31;
    int tf4 = F >> 6;                          // 64-wide col strips
    int ntiles = (M >> 4) * tf4;
    if (wid >= ntiles) return;                 // wave-uniform: EXEC stays all-ones
    int row0 = (wid / tf4) << 4;
    int ct0  = (wid % tf4) << 2;               // first 16-col tile index
    int ktiles = Kdim >> 5;

    const v16bf* B16 = (const v16bf*)Bp;
    int m = lane & 15, half = lane >> 4;
    const bf16_t* arow = A + (size_t)(row0 + m) * lda + half * 8;

    v8f acc[4] = {};
    for (int kt = 0; kt < ktiles; ++kt) {
        v16bf af = load_a_frag16(arow + kt * 32);
#pragma unroll
        for (int j = 0; j < 4; ++j) {
            v16bf bfg = B16[((size_t)(ct0 + j) * ktiles + kt) * 32 + lane];
            acc[j] = __builtin_amdgcn_wmma_f32_16x16x32_bf16(false, af, false, bfg,
                                                             (short)0, acc[j], false, false);
        }
    }
#pragma unroll
    for (int j = 0; j < 4; ++j)
        store_tile<EPI>(acc[j], row0, (ct0 + j) << 4, lane, bias, Cf, Cb, ldc);
}

// Semantic GEMM: wlog[row] += sum_col tanh(z@W1 + b1)[row,col] * w2[col]
__global__ void sem_gemm_kernel(const bf16_t* __restrict__ A,   // NT x C
                                const bf16_t* __restrict__ Bp,  // packed C x HID
                                const float* __restrict__ b1,   // HID
                                const float* __restrict__ w2,   // HID
                                float* __restrict__ wlog, int M) {
    int wid  = (int)((blockIdx.x * blockDim.x + threadIdx.x) >> 5);
    int lane = threadIdx.x & 31;
    int tf4 = HIDD >> 6;
    int ntiles = (M >> 4) * tf4;
    if (wid >= ntiles) return;
    int row0 = (wid / tf4) << 4;
    int ct0  = (wid % tf4) << 2;
    int ktiles = CC >> 5;

    const v16bf* B16 = (const v16bf*)Bp;
    int m = lane & 15, half = lane >> 4;
    const bf16_t* arow = A + (size_t)(row0 + m) * CC + half * 8;

    v8f acc[4] = {};
    for (int kt = 0; kt < ktiles; ++kt) {
        v16bf af = load_a_frag16(arow + kt * 32);
#pragma unroll
        for (int j = 0; j < 4; ++j) {
            v16bf bfg = B16[((size_t)(ct0 + j) * ktiles + kt) * 32 + lane];
            acc[j] = __builtin_amdgcn_wmma_f32_16x16x32_bf16(false, af, false, bfg,
                                                             (short)0, acc[j], false, false);
        }
    }
    float rowsum[8];
#pragma unroll
    for (int j = 0; j < 8; ++j) rowsum[j] = 0.f;
#pragma unroll
    for (int j = 0; j < 4; ++j) {
        int col = ((ct0 + j) << 4) + m;
        float b1c = b1[col], w2c = w2[col];
#pragma unroll
        for (int r = 0; r < 8; ++r) rowsum[r] += tanhf(acc[j][r] + b1c) * w2c;
    }
#pragma unroll
    for (int r = 0; r < 8; ++r) {
        float v = rowsum[r];
        v += __shfl_xor(v, 1, 32);
        v += __shfl_xor(v, 2, 32);
        v += __shfl_xor(v, 4, 32);
        v += __shfl_xor(v, 8, 32);
        if (m == 0) atomicAdd(&wlog[row0 + r + half * 8], v);
    }
}

// ---------------- edge-scatter attention passes ----------------
__global__ void edge_logits_max_kernel(const int* __restrict__ src, const int* __restrict__ dst,
                                       const bf16_t* __restrict__ Q, const bf16_t* __restrict__ Kmat,
                                       const bf16_t* __restrict__ V,
                                       float* __restrict__ alpha, unsigned* __restrict__ menc, int E) {
    int e = (int)((blockIdx.x * blockDim.x + threadIdx.x) >> 5);
    if (e >= E) return;
    int lane = threadIdx.x & 31;
    int h = lane >> 4, sub = lane & 15;
    int s = src[e], d = dst[e];
    if (lane == 0) __builtin_prefetch(V + (size_t)s * HCC, 0, 1);  // warm L2 for pass 3
    const bf16_t* q = Q    + (size_t)d * HCC + h * CC + sub * 8;
    const bf16_t* k = Kmat + (size_t)s * HCC + h * CC + sub * 8;
    float acc = 0.f;
#pragma unroll
    for (int i = 0; i < 8; ++i) acc += (float)q[i] * (float)k[i];
    acc += __shfl_xor(acc, 1, 32);
    acc += __shfl_xor(acc, 2, 32);
    acc += __shfl_xor(acc, 4, 32);
    acc += __shfl_xor(acc, 8, 32);
    if (sub == 0) {
        float a = acc * 0.08838834764831845f;   // 1/sqrt(128)
        alpha[(size_t)e * HH + h] = a;
        atomicMax(&menc[(size_t)d * HH + h], fenc(a));
    }
}

__global__ void edge_exp_sum_kernel(const int* __restrict__ dst, float* __restrict__ alpha,
                                    const unsigned* __restrict__ menc, float* __restrict__ den, long EHt) {
    long t = (long)blockIdx.x * blockDim.x + threadIdx.x;
    if (t >= EHt) return;
    int e = (int)(t >> 1), h = (int)(t & 1);
    int d = dst[e];
    float mx = fdec(menc[(size_t)d * HH + h]);
    float ex = __expf(alpha[t] - mx);
    alpha[t] = ex;
    atomicAdd(&den[(size_t)d * HH + h], ex);
}

__global__ void edge_aggregate_kernel(const int* __restrict__ src, const int* __restrict__ dst,
                                      const float* __restrict__ ex, const float* __restrict__ den,
                                      const bf16_t* __restrict__ V, float* __restrict__ u, int E) {
    int e = (int)((blockIdx.x * blockDim.x + threadIdx.x) >> 5);
    if (e >= E) return;
    int lane = threadIdx.x & 31;
    int s = src[e], d = dst[e];
    float w0 = ex[(size_t)e * HH + 0] / den[(size_t)d * HH + 0] * 0.5f;
    float w1 = ex[(size_t)e * HH + 1] / den[(size_t)d * HH + 1] * 0.5f;
    const bf16_t* v = V + (size_t)s * HCC;
    float* ud = u + (size_t)d * CC;
#pragma unroll
    for (int i = 0; i < CC / 32; ++i) {
        int c = lane + i * 32;
        float val = w0 * (float)v[c] + w1 * (float)v[CC + c];
        atomicAdd(&ud[c], val);
    }
}

// ---------------- elementwise fusion kernels ----------------
__global__ void build_cat_kernel(const float* __restrict__ u, const bf16_t* __restrict__ Xbf,
                                 bf16_t* __restrict__ cat, long total /* = N*C */) {
    long i = (long)blockIdx.x * blockDim.x + threadIdx.x;
    if (i >= total) return;
    long n = i / CC, c = i % CC;
    cat[n * (DD + CC) + c]      = (bf16_t)u[i];
    cat[n * (DD + CC) + CC + c] = Xbf[n * DD + c];
}

__global__ void fuse_store_kernel(const float* __restrict__ u, const float* __restrict__ gate,
                                  const float* __restrict__ X, float* __restrict__ embeds,
                                  bf16_t* __restrict__ zbf, int t_idx, long total) {
    long i = (long)blockIdx.x * blockDim.x + threadIdx.x;
    if (i >= total) return;
    long n = i / CC, c = i % CC;
    float a = gate[i];
    float e = tanhf(u[i]) * a + X[i] * (1.f - a);
    size_t zi = ((size_t)n * TT + t_idx) * CC + c;
    embeds[zi] = e;
    zbf[zi] = (bf16_t)e;
}

__global__ void reduce_wlog_kernel(const float* __restrict__ wlog, float* __restrict__ tsum, int M) {
    float acc[TT] = {0.f, 0.f, 0.f};
    for (long i = (long)blockIdx.x * blockDim.x + threadIdx.x; i < M;
         i += (long)gridDim.x * blockDim.x)
        acc[i % TT] += wlog[i];
#pragma unroll
    for (int t = 0; t < TT; ++t) atomicAdd(&tsum[t], acc[t]);
}

__global__ void beta_coef_kernel(const float* __restrict__ tsum, float* __restrict__ coef) {
    if (threadIdx.x != 0 || blockIdx.x != 0) return;
    float v[TT], mx = -1e30f;
#pragma unroll
    for (int t = 0; t < TT; ++t) { v[t] = tsum[t] / (float)NN; mx = fmaxf(mx, v[t]); }
    float sum = 0.f;
#pragma unroll
    for (int t = 0; t < TT; ++t) { v[t] = __expf(v[t] - mx); sum += v[t]; }
#pragma unroll
    for (int t = 0; t < TT; ++t) coef[t] += v[t] / sum / (float)SS;
}

__global__ void final_out_kernel(const float* __restrict__ embeds, const float* __restrict__ coef,
                                 float* __restrict__ out, long total) {
    long i = (long)blockIdx.x * blockDim.x + threadIdx.x;
    if (i >= total) return;
    long n = i / CC, c = i % CC;
    float acc = 0.f;
#pragma unroll
    for (int t = 0; t < TT; ++t) acc += coef[t] * embeds[((size_t)n * TT + t) * CC + c];
    out[i] = acc;
}

// ---------------- host launcher ----------------
static inline int blocks_for(long n, int bs) { return (int)((n + bs - 1) / bs); }

extern "C" void kernel_launch(void* const* d_in, const int* in_sizes, int n_in,
                              void* d_out, int out_size, void* d_ws, size_t ws_size,
                              hipStream_t stream) {
    const float* features = (const float*)d_in[0];
    const int*   edge_idx = (const int*)d_in[1];
    const float* Wq = (const float*)d_in[2];  const float* bq = (const float*)d_in[3];
    const float* Wk = (const float*)d_in[4];  const float* bk = (const float*)d_in[5];
    const float* Wv = (const float*)d_in[6];  const float* bv = (const float*)d_in[7];
    const float* Ws = (const float*)d_in[8];  const float* bs = (const float*)d_in[9];
    const float* Wg = (const float*)d_in[10]; const float* bg = (const float*)d_in[11];
    const float* W1 = (const float*)d_in[12]; const float* b1 = (const float*)d_in[13];
    const float* w2 = (const float*)d_in[14];
    float* out = (float*)d_out;

    // --- workspace carve (all buffers fully rewritten every call) ---
    char* w = (char*)d_ws;
    auto carve = [&](size_t bytes) { char* p = w; w += (bytes + 255) & ~(size_t)255; return p; };
    bf16_t* Xbf   = (bf16_t*)carve((size_t)NN * DD * 2);
    bf16_t* Wqbf  = (bf16_t*)carve((size_t)TT * DD * HCC * 2);   // packed fragments
    bf16_t* Wkbf  = (bf16_t*)carve((size_t)TT * DD * HCC * 2);
    bf16_t* Wvbf  = (bf16_t*)carve((size_t)TT * DD * HCC * 2);
    bf16_t* Wsbf  = (bf16_t*)carve((size_t)TT * DD * CC * 2);
    bf16_t* Wgbf  = (bf16_t*)carve((size_t)(DD + CC) * DD * 2);
    bf16_t* W1bf  = (bf16_t*)carve((size_t)SS * CC * HIDD * 2);
    bf16_t* Qbf   = (bf16_t*)carve((size_t)NN * HCC * 2);
    bf16_t* Kbf   = (bf16_t*)carve((size_t)NN * HCC * 2);
    bf16_t* Vbf   = (bf16_t*)carve((size_t)NN * HCC * 2);
    float*  alpha = (float*)carve((size_t)EE * HH * 4);
    unsigned* menc = (unsigned*)carve((size_t)NN * HH * 4);
    float*  den   = (float*)carve((size_t)NN * HH * 4);
    float*  u     = (float*)carve((size_t)NN * CC * 4);
    bf16_t* catbf = (bf16_t*)carve((size_t)NN * (DD + CC) * 2);
    float*  gate  = (float*)carve((size_t)NN * DD * 4);
    float*  emb   = (float*)carve((size_t)NT * CC * 4);
    bf16_t* zbf   = (bf16_t*)carve((size_t)NT * CC * 2);
    float*  wlog  = (float*)carve((size_t)NT * 4);
    float*  tsum  = (float*)carve((size_t)TT * 4);
    float*  coef  = (float*)carve((size_t)TT * 4);

    const int BS = 256;
    const int WPB = BS / 32;  // waves per block

    // --- A-side bf16 conversion + B-side fragment repack ---
    cvt_bf16_kernel<<<blocks_for((long)NN * DD, BS), BS, 0, stream>>>(features, Xbf, (long)NN * DD);
    for (int t = 0; t < TT; ++t) {
        repack_b_kernel<<<blocks_for((long)DD * HCC, BS), BS, 0, stream>>>(
            Wq + (size_t)t * DD * HCC, Wqbf + (size_t)t * DD * HCC, DD, HCC);
        repack_b_kernel<<<blocks_for((long)DD * HCC, BS), BS, 0, stream>>>(
            Wk + (size_t)t * DD * HCC, Wkbf + (size_t)t * DD * HCC, DD, HCC);
        repack_b_kernel<<<blocks_for((long)DD * HCC, BS), BS, 0, stream>>>(
            Wv + (size_t)t * DD * HCC, Wvbf + (size_t)t * DD * HCC, DD, HCC);
        repack_b_kernel<<<blocks_for((long)DD * CC, BS), BS, 0, stream>>>(
            Ws + (size_t)t * DD * CC, Wsbf + (size_t)t * DD * CC, DD, CC);
    }
    repack_b_kernel<<<blocks_for((long)(DD + CC) * DD, BS), BS, 0, stream>>>(
        Wg, Wgbf, DD + CC, DD);
    for (int s = 0; s < SS; ++s)
        repack_b_kernel<<<blocks_for((long)CC * HIDD, BS), BS, 0, stream>>>(
            W1 + (size_t)s * CC * HIDD, W1bf + (size_t)s * CC * HIDD, CC, HIDD);
    fill_zero_kernel<<<blocks_for(TT, BS), BS, 0, stream>>>((unsigned*)coef, TT);

    for (int t = 0; t < TT; ++t) {
        const int* src = edge_idx + (size_t)t * 2 * EE;
        const int* dst = src + EE;

        // Q/K/V GEMMs (bf16 out) + skip GEMM (f32 -> initializes u)
        int tiles_qkv = (NN / 16) * (HCC / 64);
        int gb_qkv = blocks_for(tiles_qkv, WPB);
        wmma_gemm_kernel<EPI_BF16><<<gb_qkv, BS, 0, stream>>>(
            Xbf, DD, Wqbf + (size_t)t * DD * HCC, bq + (size_t)t * HCC,
            nullptr, Qbf, HCC, NN, HCC, DD);
        wmma_gemm_kernel<EPI_BF16><<<gb_qkv, BS, 0, stream>>>(
            Xbf, DD, Wkbf + (size_t)t * DD * HCC, bk + (size_t)t * HCC,
            nullptr, Kbf, HCC, NN, HCC, DD);
        wmma_gemm_kernel<EPI_BF16><<<gb_qkv, BS, 0, stream>>>(
            Xbf, DD, Wvbf + (size_t)t * DD * HCC, bv + (size_t)t * HCC,
            nullptr, Vbf, HCC, NN, HCC, DD);
        int tiles_skip = (NN / 16) * (CC / 64);
        wmma_gemm_kernel<EPI_F32><<<blocks_for(tiles_skip, WPB), BS, 0, stream>>>(
            Xbf, DD, Wsbf + (size_t)t * DD * CC, bs + (size_t)t * CC,
            u, nullptr, CC, NN, CC, DD);

        // segment softmax state
        fill_zero_kernel<<<blocks_for((long)NN * HH, BS), BS, 0, stream>>>(menc, (long)NN * HH);
        fill_zero_kernel<<<blocks_for((long)NN * HH, BS), BS, 0, stream>>>((unsigned*)den, (long)NN * HH);

        // edge passes (wave per edge)
        edge_logits_max_kernel<<<blocks_for((long)EE * 32, BS), BS, 0, stream>>>(
            src, dst, Qbf, Kbf, Vbf, alpha, menc, EE);
        edge_exp_sum_kernel<<<blocks_for((long)EE * HH, BS), BS, 0, stream>>>(
            dst, alpha, menc, den, (long)EE * HH);
        edge_aggregate_kernel<<<blocks_for((long)EE * 32, BS), BS, 0, stream>>>(
            src, dst, alpha, den, Vbf, u, EE);

        // gated fusion
        build_cat_kernel<<<blocks_for((long)NN * CC, BS), BS, 0, stream>>>(
            u, Xbf, catbf, (long)NN * CC);
        int tiles_gate = (NN / 16) * (DD / 64);
        wmma_gemm_kernel<EPI_SIGMOID><<<blocks_for(tiles_gate, WPB), BS, 0, stream>>>(
            catbf, DD + CC, Wgbf, bg, gate, nullptr, DD, NN, DD, DD + CC);
        fuse_store_kernel<<<blocks_for((long)NN * CC, BS), BS, 0, stream>>>(
            u, gate, features, emb, zbf, t, (long)NN * CC);
    }

    // semantic attention pooling over edge types
    for (int s = 0; s < SS; ++s) {
        fill_zero_kernel<<<blocks_for((long)NT, BS), BS, 0, stream>>>((unsigned*)wlog, (long)NT);
        fill_zero_kernel<<<blocks_for(TT, BS), BS, 0, stream>>>((unsigned*)tsum, TT);
        int tiles_sem = (NT / 16) * (HIDD / 64);
        sem_gemm_kernel<<<blocks_for(tiles_sem, WPB), BS, 0, stream>>>(
            zbf, W1bf + (size_t)s * CC * HIDD, b1 + (size_t)s * HIDD,
            w2 + (size_t)s * HIDD, wlog, NT);
        reduce_wlog_kernel<<<64, BS, 0, stream>>>(wlog, tsum, NT);
        beta_coef_kernel<<<1, 32, 0, stream>>>(tsum, coef);
    }

    final_out_kernel<<<blocks_for((long)NN * CC, BS), BS, 0, stream>>>(
        emb, coef, out, (long)NN * CC);
}
